// DY_Block_8211977470736
// MI455X (gfx1250) — matrix operand order repaired and checked by reading, not compiled, not run on hardware
//
#include <hip/hip_runtime.h>
#include <hip/hip_bf16.h>

// ---------------- problem constants ----------------
constexpr int B_    = 32;
constexpr int CIN_  = 80;
constexpr int CEXP_ = 480;
constexpr int COUT_ = 80;
constexpr int F_    = 32;
constexpr int T_    = 125;
constexpr int N_    = F_ * T_;   // 4000
constexpr int NPAD_ = 4096;      // padded N for guard-free tiles
constexpr int KP_   = 96;        // padded K (80 -> 96) for 32-wide WMMA chunks
constexpr int CTX_  = 120;
constexpr int S_    = F_ + T_;   // 157
constexpr float EPS_ = 1e-3f;
constexpr float TEMP_ = 30.0f;

typedef __attribute__((ext_vector_type(16))) __bf16 v16bf;
typedef __attribute__((ext_vector_type(8)))  __bf16 v8bf;
typedef __attribute__((ext_vector_type(8)))  float  v8f;
typedef __attribute__((ext_vector_type(4)))  int    v4i_;

__device__ __forceinline__ float hswishf(float x) {
    float c = fminf(fmaxf(x + 3.0f, 0.0f), 6.0f);
    return x * c * (1.0f / 6.0f);
}
__device__ __forceinline__ float sigmoidf_(float x) {
    return 1.0f / (1.0f + __expf(-x));
}

__device__ __forceinline__ v8f wmma_bf16(v16bf a, v16bf b, v8f c) {
    return __builtin_amdgcn_wmma_f32_16x16x32_bf16(false, a, false, b, (short)0, c, false, false);
}

// A fragment (16x32 bf16) from LDS row-major [m][stride] tile:
// half h -> K = (h<8 ? h : h+8) + 8*(lane>>4), row M = lane&15
__device__ __forceinline__ v16bf load_a_frag(const __bf16* rowptr, int hi8) {
    v8bf lo = *(const v8bf*)(rowptr + hi8);
    v8bf hi = *(const v8bf*)(rowptr + 16 + hi8);
    return __builtin_shufflevector(lo, hi, 0,1,2,3,4,5,6,7,8,9,10,11,12,13,14,15);
}

// -------- async global->LDS 16B copy (CDNA5 GLOBAL_LOAD_ASYNC_TO_LDS_B128) --------
#if defined(__gfx1250__) && __has_builtin(__builtin_amdgcn_global_load_async_to_lds_b128)
#define HAS_ASYNC_LDS 1
#else
#define HAS_ASYNC_LDS 0
#endif

#if HAS_ASYNC_LDS
typedef __attribute__((address_space(1))) v4i_ gv4i_t;   // global int4
typedef __attribute__((address_space(3))) v4i_ lv4i_t;   // LDS int4
#endif

__device__ __forceinline__ void async_copy16(const void* g, void* l) {
#if HAS_ASYNC_LDS
    gv4i_t* gp = (gv4i_t*)(uintptr_t)g;
    lv4i_t* lp = (lv4i_t*)(uintptr_t)l;
    __builtin_amdgcn_global_load_async_to_lds_b128(gp, lp, 0, 0);
#else
    *(int4*)l = *(const int4*)g;
#endif
}

__device__ __forceinline__ void async_wait0() {
#if HAS_ASYNC_LDS
#if __has_builtin(__builtin_amdgcn_s_wait_asynccnt)
    __builtin_amdgcn_s_wait_asynccnt(0);
#else
    asm volatile("s_wait_asynccnt 0x0" ::: "memory");
#endif
#endif
}

// ---------------- K0: x0 -> transposed padded bf16 xbT[b][4096][96] ----------------
__global__ __launch_bounds__(256) void k0_xbt(const float* __restrict__ x0,
                                              __bf16* __restrict__ xbt) {
    int nb = blockIdx.x;   // 0..15
    int b  = blockIdx.y;
    int tid = threadIdx.x;
    int n0 = nb * 256;
    __shared__ __bf16 sT[256][KP_];
    int n = n0 + tid;
    for (int c = 0; c < KP_; ++c) {
        float v = (c < CIN_ && n < N_) ? x0[((size_t)b * CIN_ + c) * N_ + n] : 0.0f;
        sT[tid][c] = (__bf16)v;
    }
    __syncthreads();
    const unsigned* s = (const unsigned*)&sT[0][0];
    unsigned* d = (unsigned*)(xbt + ((size_t)b * NPAD_ + n0) * KP_);
    for (int i = tid; i < 256 * (KP_ / 2); i += 256) d[i] = s[i];
}

// ---------------- K1: channel means ----------------
__global__ __launch_bounds__(128) void k1_means(const float* __restrict__ x,
                                                float* __restrict__ cf,
                                                float* __restrict__ ct) {
    int c = blockIdx.x, b = blockIdx.y, tid = threadIdx.x;
    const float* p = x + ((size_t)b * CIN_ + c) * N_;
    if (tid < T_) {
        float s = 0.f;
        for (int f = 0; f < F_; ++f) s += p[f * T_ + tid];
        ct[((size_t)b * CIN_ + c) * T_ + tid] = s * (1.0f / F_);
    }
    if (tid < F_) {
        float s = 0.f;
        const float* row = p + tid * T_;
        for (int t = 0; t < T_; ++t) s += row[t];
        cf[((size_t)b * CIN_ + c) * F_ + tid] = s * (1.0f / T_);
    }
}

// ---------------- K2: gcat = hswish(BN(jointW @ [cf,ct])) ----------------
__global__ __launch_bounds__(256) void k2_gcat(const float* __restrict__ cf,
                                               const float* __restrict__ ct,
                                               const float* __restrict__ jw,
                                               const float* __restrict__ jg,
                                               const float* __restrict__ jb,
                                               const float* __restrict__ jm,
                                               const float* __restrict__ jv,
                                               float* __restrict__ gcat) {
    int b = blockIdx.x, tid = threadIdx.x;
    __shared__ float sIn[CIN_][S_];
    for (int idx = tid; idx < CIN_ * S_; idx += 256) {
        int c = idx / S_, s = idx % S_;
        sIn[c][s] = (s < F_) ? cf[((size_t)b * CIN_ + c) * F_ + s]
                             : ct[((size_t)b * CIN_ + c) * T_ + (s - F_)];
    }
    __syncthreads();
    for (int idx = tid; idx < CTX_ * S_; idx += 256) {
        int o = idx / S_, s = idx % S_;
        const float* w = jw + (size_t)o * CIN_;
        float acc = 0.f;
        for (int c = 0; c < CIN_; ++c) acc += w[c] * sIn[c][s];
        float inv = jg[o] * rsqrtf(jv[o] + EPS_);
        float y = acc * inv + (jb[o] - jm[o] * inv);
        gcat[((size_t)b * CTX_ + o) * S_ + s] = hswishf(y);
    }
}

// ---------------- K3: g_c, attentions, dynamic weights ----------------
__global__ __launch_bounds__(256) void k3_ctx(const float* __restrict__ gcat,
                                              const float* __restrict__ erw, const float* __restrict__ erb,
                                              const float* __restrict__ drw, const float* __restrict__ drb,
                                              const float* __restrict__ prw, const float* __restrict__ prb,
                                              const float* __restrict__ dr_w, const float* __restrict__ dr_b,
                                              const float* __restrict__ expw,
                                              const float* __restrict__ depw,
                                              const float* __restrict__ projw,
                                              float* __restrict__ coefs,
                                              float* __restrict__ wd,
                                              __bf16* __restrict__ wexp,   // [CEXP][KP_] padded
                                              __bf16* __restrict__ wp) {   // [COUT][CEXP]
    int b = blockIdx.x, tid = threadIdx.x;
    __shared__ float sGc[CTX_];
    __shared__ float sLog[12];
    __shared__ float sAtt[3][4];
    const float* g = gcat + (size_t)b * CTX_ * S_;
    if (tid < CTX_) {
        float s = 0.f;
        const float* row = g + tid * S_;
        for (int i = 0; i < S_; ++i) s += row[i];
        sGc[tid] = s * (1.0f / S_);
    }
    __syncthreads();
    if (tid < 12) {
        int j = tid >> 2, k = tid & 3;
        const float* w = (j == 0 ? erw : (j == 1 ? drw : prw)) + k * CTX_;
        float bias = (j == 0 ? erb : (j == 1 ? drb : prb))[k];
        float s = bias;
        for (int o = 0; o < CTX_; ++o) s += w[o] * sGc[o];
        sLog[tid] = s * (1.0f / TEMP_);
    }
    __syncthreads();
    if (tid < 3) {
        float m = -1e30f;
        for (int k = 0; k < 4; ++k) m = fmaxf(m, sLog[tid * 4 + k]);
        float e[4], sum = 0.f;
        for (int k = 0; k < 4; ++k) { e[k] = __expf(sLog[tid * 4 + k] - m); sum += e[k]; }
        float r = 1.0f / sum;
        for (int k = 0; k < 4; ++k) sAtt[tid][k] = e[k] * r;
    }
    __syncthreads();
    // dyn-ReLU coefs
    for (int idx = tid; idx < CEXP_ * 4; idx += 256) {
        const float* wrow = dr_w + (size_t)idx * CTX_;
        float s = dr_b[idx];
        for (int o = 0; o < CTX_; ++o) s += wrow[o] * sGc[o];
        float th = 2.0f * sigmoidf_(s) - 1.0f;
        int j = idx & 3;
        float lam = (j < 2) ? 1.0f : 0.5f;
        float iv  = (j == 0) ? 1.0f : 0.0f;
        coefs[(size_t)b * CEXP_ * 4 + idx] = th * lam + iv;
    }
    float e0 = sAtt[0][0], e1 = sAtt[0][1], e2 = sAtt[0][2], e3 = sAtt[0][3];
    float d0 = sAtt[1][0], d1 = sAtt[1][1], d2 = sAtt[1][2], d3 = sAtt[1][3];
    float p0 = sAtt[2][0], p1 = sAtt[2][1], p2 = sAtt[2][2], p3 = sAtt[2][3];
    constexpr int WE = CEXP_ * CIN_;   // 38400
    for (int idx = tid; idx < CEXP_ * KP_; idx += 256) {   // K-padded output
        int o = idx / KP_, i = idx % KP_;
        float s = 0.0f;
        if (i < CIN_) {
            int id = o * CIN_ + i;
            s = e0 * expw[id] + e1 * expw[WE + id] + e2 * expw[2 * WE + id] + e3 * expw[3 * WE + id];
        }
        wexp[(size_t)b * CEXP_ * KP_ + idx] = (__bf16)s;
    }
    constexpr int WD = CEXP_ * 9;      // 4320
    for (int idx = tid; idx < WD; idx += 256) {
        float s = d0 * depw[idx] + d1 * depw[WD + idx] + d2 * depw[2 * WD + idx] + d3 * depw[3 * WD + idx];
        wd[(size_t)b * WD + idx] = s;
    }
    constexpr int WP = COUT_ * CEXP_;  // 38400
    for (int idx = tid; idx < WP; idx += 256) {
        float s = p0 * projw[idx] + p1 * projw[WP + idx] + p2 * projw[2 * WP + idx] + p3 * projw[3 * WP + idx];
        wp[(size_t)b * WP + idx] = (__bf16)s;
    }
}

// ---------------- K4: gate sigmoids g_cf, g_ct ----------------
__global__ __launch_bounds__(256) void k4_gates(const float* __restrict__ gcat,
                                                const float* __restrict__ cfw, const float* __restrict__ cfb,
                                                const float* __restrict__ ctw, const float* __restrict__ ctb,
                                                float* __restrict__ gcf, float* __restrict__ gct) {
    int cc = blockIdx.x, b = blockIdx.y, tid = threadIdx.x;
    const float* g = gcat + (size_t)b * CTX_ * S_;
    for (int idx = tid; idx < 60 * S_; idx += 256) {
        int ol = idx / S_, s = idx % S_;
        int o = cc * 60 + ol;
        bool isF = (s < F_);
        const float* W = (isF ? cfw : ctw) + (size_t)o * CTX_;
        float acc = (isF ? cfb : ctb)[o];
        for (int c = 0; c < CTX_; ++c) acc += W[c] * g[c * S_ + s];
        float sv = sigmoidf_(acc);
        if (isF) gcf[((size_t)b * CEXP_ + o) * F_ + s] = sv;
        else     gct[((size_t)b * CEXP_ + o) * T_ + (s - F_)] = sv;
    }
}

// ---------------- K5: expand GEMM (bf16 WMMA, async LDS staging) + BN + hswish ----------------
// grid (15 Mblocks of 32, 16 Nblocks of 256, B)
__global__ __launch_bounds__(256) void k5_expand(const __bf16* __restrict__ xbt,  // [B][4096][96]
                                                 const __bf16* __restrict__ wexp, // [B][480][96]
                                                 const float* __restrict__ eg, const float* __restrict__ eb,
                                                 const float* __restrict__ em, const float* __restrict__ ev,
                                                 __bf16* __restrict__ xe) {
    const int mb = blockIdx.x, nb = blockIdx.y, b = blockIdx.z;
    const int tid = threadIdx.x;
    const int n0 = nb * 256;
    const int m0g = mb * 32;

    __shared__ __align__(64) __bf16 sA[32][KP_];
    __shared__ __align__(64) __bf16 sB[256][KP_];

    // async DMA: both tiles are byte-identical contiguous copies
    {
        const char* srcB = (const char*)(xbt + ((size_t)b * NPAD_ + n0) * KP_);
        char* dstB = (char*)&sB[0][0];
        for (int i = tid; i < 256 * KP_ / 8; i += 256)         // 3072 x 16B
            async_copy16(srcB + i * 16, dstB + i * 16);
        const char* srcA = (const char*)(wexp + (size_t)(b * CEXP_ + m0g) * KP_);
        char* dstA = (char*)&sA[0][0];
        for (int i = tid; i < 32 * KP_ / 8; i += 256)          // 384 x 16B
            async_copy16(srcA + i * 16, dstA + i * 16);
    }
    async_wait0();
    __syncthreads();

    const int wave = tid >> 5, lane = tid & 31;
    const int lr = lane & 15, hi = lane >> 4, hi8 = hi * 8, hi16 = hi * 16;

    v8f acc[2][2];
    v8f zero = {0.f, 0.f, 0.f, 0.f, 0.f, 0.f, 0.f, 0.f};
    acc[0][0] = zero; acc[0][1] = zero; acc[1][0] = zero; acc[1][1] = zero;

#pragma unroll
    for (int kc = 0; kc < KP_; kc += 32) {
        v16bf af[2];
#pragma unroll
        for (int mt = 0; mt < 2; ++mt)
            af[mt] = load_a_frag(&sA[mt * 16 + lr][kc], hi8);
#pragma unroll
        for (int nt = 0; nt < 2; ++nt) {
            int n = (wave * 2 + nt) * 16 + lr;
            v16bf bf = load_a_frag(&sB[n][kc], hi8);
#pragma unroll
            for (int mt = 0; mt < 2; ++mt)
                acc[mt][nt] = wmma_bf16(af[mt], bf, acc[mt][nt]);
        }
    }

#pragma unroll
    for (int mt = 0; mt < 2; ++mt) {
#pragma unroll
        for (int r = 0; r < 8; ++r) {
            int mg = m0g + mt * 16 + hi8 + r;
            float inv = eg[mg] * rsqrtf(ev[mg] + EPS_);
            float sh = eb[mg] - em[mg] * inv;
#pragma unroll
            for (int nt = 0; nt < 2; ++nt) {
                int ng = n0 + (wave * 2 + nt) * 16 + lr;
                float y = acc[mt][nt][r] * inv + sh;
                y = hswishf(y);
                if (ng < N_) xe[((size_t)b * CEXP_ + mg) * N_ + ng] = (__bf16)y;
            }
        }
    }
}

// ---------------- K6: fused depthwise+BN+dynReLU+gate + proj GEMM + BN + residual ----------------
// grid (16 T-tiles of 8, B)
__global__ __launch_bounds__(256) void k6_fused(const __bf16* __restrict__ xe,
                                                const __bf16* __restrict__ wp,
                                                const float* __restrict__ wd,
                                                const float* __restrict__ coefs,
                                                const float* __restrict__ gcf,
                                                const float* __restrict__ gct,
                                                const float* __restrict__ dg, const float* __restrict__ db,
                                                const float* __restrict__ dm, const float* __restrict__ dv,
                                                const float* __restrict__ pg, const float* __restrict__ pb,
                                                const float* __restrict__ pm, const float* __restrict__ pv,
                                                const float* __restrict__ x0,
                                                float* __restrict__ out) {
    const int tb = blockIdx.x, b = blockIdx.y;
    const int t0 = tb * 8;
    const int tid = threadIdx.x;
    const int wave = tid >> 5, lane = tid & 31;
    const int lr = lane & 15, hi = lane >> 4, hi8 = hi * 8, hi16 = hi * 16;

    __shared__ __align__(64) __bf16 sHalo[32][34][10];
    __shared__ __align__(64) __bf16 sAct[256][32];
    __shared__ __align__(64) __bf16 sA[80][32];

    v8f acc[5][2];
    v8f zero = {0.f, 0.f, 0.f, 0.f, 0.f, 0.f, 0.f, 0.f};
#pragma unroll
    for (int mt = 0; mt < 5; ++mt) { acc[mt][0] = zero; acc[mt][1] = zero; }

    const int cth = tid >> 3;   // 0..31 channel-in-chunk for conv stage
    const int tth = tid & 7;    // 0..7  t-in-tile for conv stage
    const __bf16* wpBase = wp + (size_t)b * COUT_ * CEXP_;

    for (int kc = 0; kc < CEXP_; kc += 32) {
        __syncthreads();
        // async DMA of A chunk of wp: 80 rows x 32 k (64B per row)
        for (int i = tid; i < 320; i += 256) {
            int m = i >> 2, q = i & 3;
            async_copy16(wpBase + (size_t)m * CEXP_ + kc + q * 8, &sA[m][q * 8]);
        }
        // stage halo tile: 32 ch x (32+2)f x (8+2)t (boundary zero-fill)
        for (int idx = tid; idx < 32 * 34 * 10; idx += 256) {
            int c = idx / 340, rem = idx % 340;
            int fh = rem / 10, th = rem % 10;
            int f = fh - 1, t = t0 - 1 + th;
            __bf16 v = (__bf16)0.0f;
            if ((unsigned)f < (unsigned)F_ && (unsigned)t < (unsigned)T_)
                v = xe[((size_t)b * CEXP_ + kc + c) * N_ + f * T_ + t];
            sHalo[c][fh][th] = v;
        }
        async_wait0();
        __syncthreads();

        // depthwise conv + dep BN + dynReLU + gating -> sAct (bf16, [pos][k])
        {
            int cg = kc + cth;
            const float* w9 = &wd[((size_t)b * CEXP_ + cg) * 9];
            float w0 = w9[0], w1 = w9[1], w2 = w9[2];
            float w3 = w9[3], w4 = w9[4], w5 = w9[5];
            float w6 = w9[6], w7 = w9[7], w8 = w9[8];
            float invd = dg[cg] * rsqrtf(dv[cg] + EPS_);
            float shd = db[cg] - dm[cg] * invd;
            const float* c4 = &coefs[((size_t)b * CEXP_ + cg) * 4];
            float a0 = c4[0], a1 = c4[1], b0 = c4[2], b1 = c4[3];
            int t = t0 + tth;
            float sgt = (t < T_) ? gct[((size_t)b * CEXP_ + cg) * T_ + t] : 0.0f;
            const float* gcfp = &gcf[((size_t)b * CEXP_ + cg) * F_];
            for (int f = 0; f < F_; ++f) {
                float s =
                    w0 * (float)sHalo[cth][f + 0][tth + 0] + w1 * (float)sHalo[cth][f + 0][tth + 1] + w2 * (float)sHalo[cth][f + 0][tth + 2] +
                    w3 * (float)sHalo[cth][f + 1][tth + 0] + w4 * (float)sHalo[cth][f + 1][tth + 1] + w5 * (float)sHalo[cth][f + 1][tth + 2] +
                    w6 * (float)sHalo[cth][f + 2][tth + 0] + w7 * (float)sHalo[cth][f + 2][tth + 1] + w8 * (float)sHalo[cth][f + 2][tth + 2];
                float y = s * invd + shd;
                y = fmaxf(fmaf(y, a0, b0), fmaf(y, a1, b1));
                y *= gcfp[f] * sgt;
                sAct[f * 8 + tth][cth] = (__bf16)y;
            }
        }
        __syncthreads();

        // WMMA: 5 Mtiles x 2 Ntiles per wave, K=32 per chunk
        v16bf af[5];
#pragma unroll
        for (int mt = 0; mt < 5; ++mt)
            af[mt] = load_a_frag(&sA[mt * 16 + lr][0], hi8);
#pragma unroll
        for (int nt = 0; nt < 2; ++nt) {
            int n = (wave * 2 + nt) * 16 + lr;
            v16bf bf = load_a_frag(&sAct[n][0], hi8);
#pragma unroll
            for (int mt = 0; mt < 5; ++mt)
                acc[mt][nt] = wmma_bf16(af[mt], bf, acc[mt][nt]);
        }
    }

    // epilogue: proj BN + residual
#pragma unroll
    for (int mt = 0; mt < 5; ++mt) {
#pragma unroll
        for (int r = 0; r < 8; ++r) {
            int m = mt * 16 + hi8 + r;
            float inv = pg[m] * rsqrtf(pv[m] + EPS_);
            float sh = pb[m] - pm[m] * inv;
#pragma unroll
            for (int nt = 0; nt < 2; ++nt) {
                int pos = (wave * 2 + nt) * 16 + lr;
                int f = pos >> 3, tt = pos & 7;
                int t = t0 + tt;
                if (t < T_) {
                    size_t o = ((size_t)b * COUT_ + m) * N_ + f * T_ + t;
                    out[o] = acc[mt][nt][r] * inv + sh + x0[o];
                }
            }
        }
    }
}

// ---------------- launch ----------------
extern "C" void kernel_launch(void* const* d_in, const int* in_sizes, int n_in,
                              void* d_out, int out_size, void* d_ws, size_t ws_size,
                              hipStream_t stream) {
    const float* x        = (const float*)d_in[0];
    const float* jw       = (const float*)d_in[1];
    const float* jg       = (const float*)d_in[2];
    const float* jb       = (const float*)d_in[3];
    const float* jm       = (const float*)d_in[4];
    const float* jv       = (const float*)d_in[5];
    const float* cfw      = (const float*)d_in[6];
    const float* cfb      = (const float*)d_in[7];
    const float* ctw      = (const float*)d_in[8];
    const float* ctb      = (const float*)d_in[9];
    const float* erw      = (const float*)d_in[10];
    const float* erb      = (const float*)d_in[11];
    const float* expw     = (const float*)d_in[12];
    const float* eg       = (const float*)d_in[13];
    const float* eb       = (const float*)d_in[14];
    const float* em       = (const float*)d_in[15];
    const float* ev       = (const float*)d_in[16];
    const float* drw      = (const float*)d_in[17];
    const float* drb      = (const float*)d_in[18];
    const float* depw     = (const float*)d_in[19];
    const float* dgp      = (const float*)d_in[20];
    const float* dbp      = (const float*)d_in[21];
    const float* dmp      = (const float*)d_in[22];
    const float* dvp      = (const float*)d_in[23];
    const float* dr_w     = (const float*)d_in[24];
    const float* dr_b     = (const float*)d_in[25];
    const float* prw      = (const float*)d_in[26];
    const float* prb      = (const float*)d_in[27];
    const float* projw    = (const float*)d_in[28];
    const float* pgp      = (const float*)d_in[29];
    const float* pbp      = (const float*)d_in[30];
    const float* pmp      = (const float*)d_in[31];
    const float* pvp      = (const float*)d_in[32];
    float* out = (float*)d_out;

    char* p = (char*)d_ws;
    auto carve = [&](size_t bytes) -> void* {
        void* r = (void*)p;
        p += (bytes + 255) & ~(size_t)255;
        return r;
    };
    float*  cf    = (float*) carve((size_t)B_ * CIN_ * F_ * 4);
    float*  ct    = (float*) carve((size_t)B_ * CIN_ * T_ * 4);
    float*  gcat  = (float*) carve((size_t)B_ * CTX_ * S_ * 4);
    float*  gcf   = (float*) carve((size_t)B_ * CEXP_ * F_ * 4);        // sigmoid(g_cf)
    float*  gct   = (float*) carve((size_t)B_ * CEXP_ * T_ * 4);        // sigmoid(g_ct)
    float*  coefs = (float*) carve((size_t)B_ * CEXP_ * 4 * 4);
    float*  wd    = (float*) carve((size_t)B_ * CEXP_ * 9 * 4);
    __bf16* wexp  = (__bf16*)carve((size_t)B_ * CEXP_ * KP_ * 2);       // K-padded
    __bf16* wp    = (__bf16*)carve((size_t)B_ * COUT_ * CEXP_ * 2);
    __bf16* xbt   = (__bf16*)carve((size_t)B_ * NPAD_ * KP_ * 2);       // ~25 MB
    __bf16* xe    = (__bf16*)carve((size_t)B_ * CEXP_ * N_ * 2);        // ~123 MB

    k0_xbt<<<dim3(16, B_), 256, 0, stream>>>(x, xbt);
    k1_means<<<dim3(CIN_, B_), 128, 0, stream>>>(x, cf, ct);
    k2_gcat<<<B_, 256, 0, stream>>>(cf, ct, jw, jg, jb, jm, jv, gcat);
    k3_ctx<<<B_, 256, 0, stream>>>(gcat, erw, erb, drw, drb, prw, prb, dr_w, dr_b,
                                   expw, depw, projw, coefs, wd, wexp, wp);
    k4_gates<<<dim3(8, B_), 256, 0, stream>>>(gcat, cfw, cfb, ctw, ctb, gcf, gct);
    k5_expand<<<dim3(15, 16, B_), 256, 0, stream>>>(xbt, wexp, eg, eb, em, ev, xe);
    k6_fused<<<dim3(16, B_), 256, 0, stream>>>(xe, wp, wd, coefs, gcf, gct,
                                               dgp, dbp, dmp, dvp, pgp, pbp, pmp, pvp,
                                               x, out);
}
